// acLSTM_31533649887639
// MI455X (gfx1250) — matrix operand order repaired and checked by reading, not compiled, not run on hardware
//
#include <hip/hip_runtime.h>
#include <hip/hip_bf16.h>

typedef _Float16 v4h  __attribute__((ext_vector_type(4)));
typedef _Float16 v8h  __attribute__((ext_vector_type(8)));
typedef _Float16 v16h __attribute__((ext_vector_type(16)));
typedef float    v4f  __attribute__((ext_vector_type(4)));
typedef float    v8f  __attribute__((ext_vector_type(8)));

#define HID   1024
#define BATCH 32
#define TLEN  200
#define FRAME 231

// ---------------------------------------------------------------------------
// fragment loaders (16x16x32 f16 WMMA layouts, wave32) -- all unconditional,
// all sources padded so no bounds checks exist inside GEMM loops.
// A (16x32, M x K): lane L<16 -> row M=L, K = 0..15; lane L>=16 ->
//                   row M=L-16, K = 16..31
// B (32x16, K x N): lane L<16 -> col N=L, K = 0..15; lane L>=16 -> col N=L-16,
//                   K = 16..31 (elements linear in K)
// C/D (16x16 f32): lane L<16 -> elem j=(M=j,N=L); lane L>=16 -> (M=8+j,N=L-16)
// ---------------------------------------------------------------------------

__device__ inline v16h load_a_frag(const _Float16* __restrict__ base, long row,
                                   int stride, int kBase, bool hi) {
  const _Float16* p = base + row * stride + kBase + (hi ? 16 : 0);
  v8h lo = *(const v8h*)(p);
  v8h hv = *(const v8h*)(p + 8);
  return __builtin_shufflevector(lo, hv, 0, 1, 2, 3, 4, 5, 6, 7, 8, 9, 10, 11,
                                 12, 13, 14, 15);
}

// B fragment, f16 weights (pre-converted): 32 contiguous bytes per lane.
__device__ inline v16h load_w_frag(const _Float16* __restrict__ W, long row,
                                   int kStride, int kBase, bool hi) {
  const _Float16* p = W + row * kStride + kBase + (hi ? 16 : 0);
  v8h lo = *(const v8h*)(p);
  v8h hv = *(const v8h*)(p + 8);
  return __builtin_shufflevector(lo, hv, 0, 1, 2, 3, 4, 5, 6, 7, 8, 9, 10, 11,
                                 12, 13, 14, 15);
}

// B fragment, f32 weights (fallback path): 64 contiguous bytes + convert.
__device__ inline v16h load_w_frag(const float* __restrict__ W, long row,
                                   int kStride, int kBase, bool hi) {
  const float* p = W + row * kStride + kBase + (hi ? 16 : 0);
  v4f q0 = *(const v4f*)(p);
  v4f q1 = *(const v4f*)(p + 4);
  v4f q2 = *(const v4f*)(p + 8);
  v4f q3 = *(const v4f*)(p + 12);
  v8h a = __builtin_shufflevector(__builtin_convertvector(q0, v4h),
                                  __builtin_convertvector(q1, v4h),
                                  0, 1, 2, 3, 4, 5, 6, 7);
  v8h b = __builtin_shufflevector(__builtin_convertvector(q2, v4h),
                                  __builtin_convertvector(q3, v4h),
                                  0, 1, 2, 3, 4, 5, 6, 7);
  return __builtin_shufflevector(a, b, 0, 1, 2, 3, 4, 5, 6, 7, 8, 9, 10, 11,
                                 12, 13, 14, 15);
}

__device__ inline float sigm_f(float x) { return 1.0f / (1.0f + __expf(-x)); }
__device__ inline float tanh_f(float x) {
  return 2.0f / (1.0f + __expf(-2.0f * x)) - 1.0f;
}

// ---------------------------------------------------------------------------
// Fused LSTM layer: gates = x@Wih^T + h@Whh^T + bih + bhh, then cell update.
// 128 waves: wave = mTile(2) x hiddenTile(64); each wave owns all 4 gates for
// its 16 hidden columns -> pointwise fuses with zero cross-wave traffic.
// ---------------------------------------------------------------------------
template <typename WT, int XKT>
__global__ __launch_bounds__(256) void lstm_layer_kernel(
    const _Float16* __restrict__ xIn, int xStride,
    const WT* __restrict__ Wih, int wihStride,
    const _Float16* __restrict__ hIn,
    const WT* __restrict__ Whh,
    const float* __restrict__ bih, const float* __restrict__ bhh,
    float* __restrict__ c, _Float16* __restrict__ hOut) {
  int lane = threadIdx.x & 31;
  int wave = (blockIdx.x * blockDim.x + threadIdx.x) >> 5;  // 0..127
  int mTile = wave & 1;
  int n0 = (wave >> 1) << 4;  // hidden column base: 0..1008
  int laneN = lane & 15;
  bool hi = lane >= 16;
  long rowA = mTile * 16 + laneN;
  long rowW[4];
#pragma unroll
  for (int g = 0; g < 4; ++g) rowW[g] = (long)g * HID + n0 + laneN;

  v8f acc[4];
#pragma unroll
  for (int g = 0; g < 4; ++g)
#pragma unroll
    for (int j = 0; j < 8; ++j) acc[g][j] = 0.0f;

  // x @ Wih^T   (XKT compile-time K tiles of 32)
#pragma unroll 4
  for (int kt = 0; kt < XKT; ++kt) {
    int kBase = kt << 5;
    v16h a = load_a_frag(xIn, rowA, xStride, kBase, hi);
#pragma unroll
    for (int g = 0; g < 4; ++g) {
      v16h b = load_w_frag(Wih, rowW[g], wihStride, kBase, hi);
      acc[g] = __builtin_amdgcn_wmma_f32_16x16x32_f16(false, a, false, b,
                                                      (short)0, acc[g], false,
                                                      false);
    }
  }
  // h @ Whh^T   (32 K tiles)
#pragma unroll 4
  for (int kt = 0; kt < HID / 32; ++kt) {
    int kBase = kt << 5;
    v16h a = load_a_frag(hIn, rowA, HID, kBase, hi);
#pragma unroll
    for (int g = 0; g < 4; ++g) {
      v16h b = load_w_frag(Whh, rowW[g], HID, kBase, hi);
      acc[g] = __builtin_amdgcn_wmma_f32_16x16x32_f16(false, a, false, b,
                                                      (short)0, acc[g], false,
                                                      false);
    }
  }

  int n = n0 + laneN;
  float bI = bih[n] + bhh[n];
  float bF = bih[HID + n] + bhh[HID + n];
  float bG = bih[2 * HID + n] + bhh[2 * HID + n];
  float bO = bih[3 * HID + n] + bhh[3 * HID + n];
#pragma unroll
  for (int j = 0; j < 8; ++j) {
    int m = mTile * 16 + (hi ? 8 + j : j);
    float gi = acc[0][j] + bI;
    float gf = acc[1][j] + bF;
    float gg = acc[2][j] + bG;
    float go = acc[3][j] + bO;
    long idx = (long)m * HID + n;
    float cOld = c[idx];
    float cNew = sigm_f(gf) * cOld + sigm_f(gi) * tanh_f(gg);
    c[idx] = cNew;
    hOut[idx] = (_Float16)(sigm_f(go) * tanh_f(cNew));
  }
}

// ---------------------------------------------------------------------------
// Decoder: yPre = h3 @ decWpad^T + decb  (32 x 231, K=1024, N padded to 240).
// 30 waves.
// ---------------------------------------------------------------------------
template <typename WT>
__global__ __launch_bounds__(32) void decoder_kernel(
    const _Float16* __restrict__ h3, const WT* __restrict__ decWp,
    const float* __restrict__ decb, float* __restrict__ yPre) {
  int lane = threadIdx.x;
  int wave = blockIdx.x;  // 0..29
  int mTile = wave & 1;
  int n0 = (wave >> 1) << 4;  // 0..224
  int laneN = lane & 15;
  bool hi = lane >= 16;
  long rowA = mTile * 16 + laneN;
  int n = n0 + laneN;

  v8f acc;
#pragma unroll
  for (int j = 0; j < 8; ++j) acc[j] = 0.0f;

#pragma unroll 4
  for (int kt = 0; kt < HID / 32; ++kt) {
    int kBase = kt << 5;
    v16h a = load_a_frag(h3, rowA, HID, kBase, hi);
    v16h b = load_w_frag(decWp, (long)n, HID, kBase, hi);
    acc = __builtin_amdgcn_wmma_f32_16x16x32_f16(false, a, false, b, (short)0,
                                                 acc, false, false);
  }
  if (n < FRAME) {
    float bn = decb[n];
#pragma unroll
    for (int j = 0; j < 8; ++j) {
      int m = mTile * 16 + (hi ? 8 + j : j);
      yPre[(long)m * FRAME + n] = acc[j] + bn;
    }
  }
}

// ---------------------------------------------------------------------------
// renorm_quat: keep [0:3], L2-normalize each group of 4 over [3:231].
// ---------------------------------------------------------------------------
__global__ void renorm_kernel(const float* __restrict__ yPre,
                              float* __restrict__ out, int t) {
  int b = blockIdx.x;
  int j = threadIdx.x;  // block 64
  const float* src = yPre + (long)b * FRAME;
  float* dst = out + ((long)b * TLEN + t) * FRAME;
  if (j < 57) {
    int base = 3 + j * 4;
    float v0 = src[base + 0], v1 = src[base + 1];
    float v2 = src[base + 2], v3 = src[base + 3];
    float nrm = sqrtf(v0 * v0 + v1 * v1 + v2 * v2 + v3 * v3) + 1e-8f;
    float inv = 1.0f / nrm;
    dst[base + 0] = v0 * inv;
    dst[base + 1] = v1 * inv;
    dst[base + 2] = v2 * inv;
    dst[base + 3] = v3 * inv;
  } else if (j < 60) {
    dst[j - 57] = src[j - 57];
  }
}

// ---------------------------------------------------------------------------
// Build x_f16[32][256]: masked select of seq[:,t,:] vs previous output, padded.
// ---------------------------------------------------------------------------
__global__ void prep_kernel(const float* __restrict__ seq,
                            const float* __restrict__ out,
                            _Float16* __restrict__ x, int t) {
  int b = blockIdx.x;
  int f = threadIdx.x;  // block 256
  bool gt = (t % 10) < 5;
  float v = 0.0f;
  if (f < FRAME) {
    v = gt ? seq[((long)b * TLEN + t) * FRAME + f]
           : out[((long)b * TLEN + (t - 1)) * FRAME + f];
  }
  x[b * 256 + f] = (_Float16)v;
}

__global__ void init_state_kernel(float* __restrict__ c,
                                  _Float16* __restrict__ h, int nC, int nH) {
  int i = blockIdx.x * blockDim.x + threadIdx.x;
  if (i < nC) c[i] = 0.0f;
  if (i < nH) h[i] = (_Float16)0.0f;
}

// Pad/convert: dst[r][k] = (r<srcRows && k<srcK) ? src[r][k] : 0
template <typename WT>
__global__ void convert_pad_kernel(const float* __restrict__ src, int srcK,
                                   int srcRows, WT* __restrict__ dst, int dstK,
                                   long total) {
  long i = (long)blockIdx.x * blockDim.x + threadIdx.x;
  if (i >= total) return;
  long r = i / dstK;
  int k = (int)(i - r * dstK);
  float v = (k < srcK && r < srcRows) ? src[r * srcK + k] : 0.0f;
  dst[i] = (WT)v;
}

// ---------------------------------------------------------------------------
extern "C" void kernel_launch(void* const* d_in, const int* in_sizes, int n_in,
                              void* d_out, int out_size, void* d_ws,
                              size_t ws_size, hipStream_t stream) {
  const float* seq  = (const float*)d_in[0];
  const float* Wih1 = (const float*)d_in[1];
  const float* Whh1 = (const float*)d_in[2];
  const float* bih1 = (const float*)d_in[3];
  const float* bhh1 = (const float*)d_in[4];
  const float* Wih2 = (const float*)d_in[5];
  const float* Whh2 = (const float*)d_in[6];
  const float* bih2 = (const float*)d_in[7];
  const float* bhh2 = (const float*)d_in[8];
  const float* Wih3 = (const float*)d_in[9];
  const float* Whh3 = (const float*)d_in[10];
  const float* bih3 = (const float*)d_in[11];
  const float* bhh3 = (const float*)d_in[12];
  const float* decW = (const float*)d_in[13];
  const float* decb = (const float*)d_in[14];
  float* out = (float*)d_out;

  const long SB = (long)BATCH * HID;  // 32*1024 state slab
  char* w = (char*)d_ws;
  float* c1 = (float*)w; w += SB * sizeof(float);
  float* c2 = (float*)w; w += SB * sizeof(float);
  float* c3 = (float*)w; w += SB * sizeof(float);
  _Float16* h1 = (_Float16*)w; w += 2 * SB * sizeof(_Float16);  // double-buffer
  _Float16* h2 = (_Float16*)w; w += 2 * SB * sizeof(_Float16);
  _Float16* h3 = (_Float16*)w; w += 2 * SB * sizeof(_Float16);
  _Float16* xbuf = (_Float16*)w; w += BATCH * 256 * sizeof(_Float16);
  float* yPre = (float*)w; w += (long)BATCH * FRAME * sizeof(float);

  const long W1E = 4096L * 256;   // padded Wih1 elements
  const long WHE = 4096L * 1024;  // full-size weight elements
  const long WDE = 240L * 1024;   // padded decW elements
  size_t stateBytes = (size_t)(w - (char*)d_ws);
  size_t f16Bytes = (size_t)(W1E + 5 * WHE + WDE) * sizeof(_Float16);
  bool bigWS = ws_size >= stateBytes + f16Bytes + 256;

  // zero c1..c3 (contiguous) and h1..h3 (contiguous, both parities)
  init_state_kernel<<<(6 * (int)SB + 255) / 256, 256, 0, stream>>>(
      c1, h1, 3 * (int)SB, 6 * (int)SB);

  if (bigWS) {
    // ---- f16 pre-converted weight path (weights ~43MB, L2-resident) ----
    _Float16* Wih1p = (_Float16*)w; w += W1E * sizeof(_Float16);
    _Float16* Whh1p = (_Float16*)w; w += WHE * sizeof(_Float16);
    _Float16* Wih2p = (_Float16*)w; w += WHE * sizeof(_Float16);
    _Float16* Whh2p = (_Float16*)w; w += WHE * sizeof(_Float16);
    _Float16* Wih3p = (_Float16*)w; w += WHE * sizeof(_Float16);
    _Float16* Whh3p = (_Float16*)w; w += WHE * sizeof(_Float16);
    _Float16* decWp = (_Float16*)w; w += WDE * sizeof(_Float16);

    convert_pad_kernel<_Float16><<<(int)((W1E + 255) / 256), 256, 0, stream>>>(
        Wih1, FRAME, 4096, Wih1p, 256, W1E);
    convert_pad_kernel<_Float16><<<(int)((WHE + 255) / 256), 256, 0, stream>>>(
        Whh1, HID, 4096, Whh1p, HID, WHE);
    convert_pad_kernel<_Float16><<<(int)((WHE + 255) / 256), 256, 0, stream>>>(
        Wih2, HID, 4096, Wih2p, HID, WHE);
    convert_pad_kernel<_Float16><<<(int)((WHE + 255) / 256), 256, 0, stream>>>(
        Whh2, HID, 4096, Whh2p, HID, WHE);
    convert_pad_kernel<_Float16><<<(int)((WHE + 255) / 256), 256, 0, stream>>>(
        Wih3, HID, 4096, Wih3p, HID, WHE);
    convert_pad_kernel<_Float16><<<(int)((WHE + 255) / 256), 256, 0, stream>>>(
        Whh3, HID, 4096, Whh3p, HID, WHE);
    convert_pad_kernel<_Float16><<<(int)((WDE + 255) / 256), 256, 0, stream>>>(
        decW, HID, FRAME, decWp, HID, WDE);

    for (int t = 0; t < TLEN; ++t) {
      int p = t & 1;
      _Float16* h1i = h1 + (long)p * SB;
      _Float16* h1o = h1 + (long)(p ^ 1) * SB;
      _Float16* h2i = h2 + (long)p * SB;
      _Float16* h2o = h2 + (long)(p ^ 1) * SB;
      _Float16* h3i = h3 + (long)p * SB;
      _Float16* h3o = h3 + (long)(p ^ 1) * SB;

      prep_kernel<<<BATCH, 256, 0, stream>>>(seq, out, xbuf, t);
      lstm_layer_kernel<_Float16, 8><<<16, 256, 0, stream>>>(
          xbuf, 256, Wih1p, 256, h1i, Whh1p, bih1, bhh1, c1, h1o);
      lstm_layer_kernel<_Float16, 32><<<16, 256, 0, stream>>>(
          h1o, HID, Wih2p, HID, h2i, Whh2p, bih2, bhh2, c2, h2o);
      lstm_layer_kernel<_Float16, 32><<<16, 256, 0, stream>>>(
          h2o, HID, Wih3p, HID, h3i, Whh3p, bih3, bhh3, c3, h3o);
      decoder_kernel<_Float16><<<30, 32, 0, stream>>>(h3o, decWp, decb, yPre);
      renorm_kernel<<<BATCH, 64, 0, stream>>>(yPre, out, t);
    }
  } else {
    // ---- small-ws fallback: f32 weights, convert in-register; only the
    // ragged matrices get padded f32 copies (~5MB) ----
    float* Wih1p = (float*)w; w += W1E * sizeof(float);
    float* decWp = (float*)w; w += WDE * sizeof(float);

    convert_pad_kernel<float><<<(int)((W1E + 255) / 256), 256, 0, stream>>>(
        Wih1, FRAME, 4096, Wih1p, 256, W1E);
    convert_pad_kernel<float><<<(int)((WDE + 255) / 256), 256, 0, stream>>>(
        decW, HID, FRAME, decWp, HID, WDE);

    for (int t = 0; t < TLEN; ++t) {
      int p = t & 1;
      _Float16* h1i = h1 + (long)p * SB;
      _Float16* h1o = h1 + (long)(p ^ 1) * SB;
      _Float16* h2i = h2 + (long)p * SB;
      _Float16* h2o = h2 + (long)(p ^ 1) * SB;
      _Float16* h3i = h3 + (long)p * SB;
      _Float16* h3o = h3 + (long)(p ^ 1) * SB;

      prep_kernel<<<BATCH, 256, 0, stream>>>(seq, out, xbuf, t);
      lstm_layer_kernel<float, 8><<<16, 256, 0, stream>>>(
          xbuf, 256, Wih1p, 256, h1i, Whh1, bih1, bhh1, c1, h1o);
      lstm_layer_kernel<float, 32><<<16, 256, 0, stream>>>(
          h1o, HID, Wih2, HID, h2i, Whh2, bih2, bhh2, c2, h2o);
      lstm_layer_kernel<float, 32><<<16, 256, 0, stream>>>(
          h2o, HID, Wih3, HID, h3i, Whh3, bih3, bhh3, c3, h3o);
      decoder_kernel<float><<<30, 32, 0, stream>>>(h3o, decWp, decb, yPre);
      renorm_kernel<<<BATCH, 64, 0, stream>>>(yPre, out, t);
    }
  }
}
